// BitFeedForward_45174466020062
// MI455X (gfx1250) — 8-bit matrix operand order repaired and checked by reading, compiled, not run on hardware
//
#include <hip/hip_runtime.h>
#include <cstdint>


typedef __attribute__((ext_vector_type(8))) int v8i;
typedef int v4i_ __attribute__((__vector_size__(4 * sizeof(int))));
typedef __attribute__((address_space(1))) v4i_ gv4i;
typedef __attribute__((address_space(3))) v4i_ lv4i;

#define TOKENS   8192
#define DDIM     2048
#define FDIM     8192
#define WELEMS   (FDIM * DDIM)
#define EPSV     1e-5f

#define BM 128
#define BN 128
#define BK 64
#define LDSTR 80

#if __has_builtin(__builtin_amdgcn_global_load_async_to_lds_b128)
#define ASYNC_LDS 1
#else
#define ASYNC_LDS 0
#endif

__device__ __forceinline__ void cp16(const int8_t* g, int8_t* l) {
#if ASYNC_LDS
    gv4i* gp = (gv4i*)(uintptr_t)g;
    lv4i* lp = (lv4i*)(uint32_t)(uintptr_t)l;
    __builtin_amdgcn_global_load_async_to_lds_b128(gp, lp, 0, 0);
#else
    *(int4*)l = *(const int4*)g;
#endif
}

template <int N>
__device__ __forceinline__ void wait_async() {
#if ASYNC_LDS
#if __has_builtin(__builtin_amdgcn_s_wait_asynccnt)
    __builtin_amdgcn_s_wait_asynccnt((unsigned short)N);
#else
    asm volatile("s_wait_asynccnt %0" ::"i"(N) : "memory");
#endif
#endif
}

__global__ void zero_scalars_kernel(float* s) {
    if (threadIdx.x < 4) s[threadIdx.x] = 0.0f;
}

__global__ __launch_bounds__(256)
void abs_sum_kernel(const float* __restrict__ w, long long n, float* accum) {
    __shared__ float red[256];
    float acc = 0.0f;
    long long stride = (long long)gridDim.x * blockDim.x;
    for (long long i = (long long)blockIdx.x * blockDim.x + threadIdx.x; i < n; i += stride)
        acc += fabsf(w[i]);
    red[threadIdx.x] = acc;
    __syncthreads();
    for (int s2 = 128; s2 > 0; s2 >>= 1) {
        if ((int)threadIdx.x < s2) red[threadIdx.x] += red[threadIdx.x + s2];
        __syncthreads();
    }
    if (threadIdx.x == 0) atomicAdd(accum, red[0]);
}

__global__ void make_scales_kernel(float* s) {
    if (threadIdx.x == 0) {
        s[2] = fmaxf(s[0] / (float)WELEMS, EPSV);
        s[3] = fmaxf(s[1] / (float)WELEMS, EPSV);
    }
}

__global__ __launch_bounds__(256)
void quant_w_kernel(const float* __restrict__ w, int8_t* __restrict__ wq,
                    long long n, const float* __restrict__ dqPtr) {
    const float scale = 1.0f / dqPtr[0];
    long long stride = (long long)gridDim.x * blockDim.x;
    for (long long i = (long long)blockIdx.x * blockDim.x + threadIdx.x; i < n; i += stride) {
        float q = rintf(w[i] * scale);
        q = fminf(fmaxf(q, -1.0f), 1.0f);
        wq[i] = (int8_t)q;
    }
}

__global__ __launch_bounds__(256)
void quant_x_kernel(const float* __restrict__ x, int8_t* __restrict__ xq,
                    float* __restrict__ dq, int Dd) {
    __shared__ float red[256];
    const int row = blockIdx.x;
    const int tid = threadIdx.x;
    const float* xr = x + (size_t)row * Dd;
    float m = 0.0f;
    for (int i = tid; i < Dd; i += 256) m = fmaxf(m, fabsf(xr[i]));
    red[tid] = m;
    __syncthreads();
    for (int s2 = 128; s2 > 0; s2 >>= 1) {
        if (tid < s2) red[tid] = fmaxf(red[tid], red[tid + s2]);
        __syncthreads();
    }
    const float mx = fmaxf(red[0], EPSV);
    const float scale = 127.0f / mx;
    for (int i = tid; i < Dd; i += 256) {
        float q = rintf(xr[i] * scale);
        q = fminf(fmaxf(q, -128.0f), 127.0f);
        xq[(size_t)row * Dd + i] = (int8_t)q;
    }
    if (tid == 0) dq[row] = mx / 127.0f;
}

template <bool GELU>
__global__ __launch_bounds__(256)
void gemm_i8_kernel(const int8_t* __restrict__ A, const int8_t* __restrict__ Bw,
                    const float* __restrict__ dqA, const float* __restrict__ scal,
                    int dqBidx, const float* __restrict__ bias,
                    float* __restrict__ C, int M, int N, int K) {
    __shared__ __align__(16) int8_t As[2][BM * LDSTR];
    __shared__ __align__(16) int8_t Bs[2][BN * LDSTR];

    const int tid   = threadIdx.x;
    const int wave  = tid >> 5;
    const int lane  = tid & 31;
    const int half  = lane >> 4;
    const int l16   = lane & 15;
    const int waveM = (wave & 1) * 64;
    const int waveN = (wave >> 1) * 32;
    const int blockM = blockIdx.y * BM;
    const int blockN = blockIdx.x * BN;

    const int lrow = tid >> 1;
    const int loff = (tid & 1) * 32;

    const float dqB = scal[dqBidx];

    v8i acc[4][2] = {};

    auto stage = [&](int k0, int buf) {
        const int8_t* ag = A + (size_t)(blockM + lrow) * K + k0 + loff;
        int8_t* as = &As[buf][lrow * LDSTR + loff];
        cp16(ag,      as);
        cp16(ag + 16, as + 16);
        const int8_t* bg = Bw + (size_t)(blockN + lrow) * K + k0 + loff;
        int8_t* bs = &Bs[buf][lrow * LDSTR + loff];
        cp16(bg,      bs);
        cp16(bg + 16, bs + 16);
    };

    auto compute = [&](int buf) {
        v8i afrag[4];
        #pragma unroll
        for (int mt = 0; mt < 4; ++mt) {
            const int8_t* base = &As[buf][(waveM + mt * 16 + l16) * LDSTR + half * 16];
            #pragma unroll
            for (int p = 0; p < 4; ++p) {
                int2 c = *(const int2*)(base + (p & 1) * 8 + (p >> 1) * 32);
                afrag[mt][2 * p]     = c.x;
                afrag[mt][2 * p + 1] = c.y;
            }
        }
        v8i bfrag[2];
        #pragma unroll
        for (int nt = 0; nt < 2; ++nt) {
            const int8_t* base = &Bs[buf][(waveN + nt * 16 + l16) * LDSTR];
            int4 c0 = *(const int4*)(base + half * 16);
            int4 c1 = *(const int4*)(base + 32 + half * 16);
            bfrag[nt][0] = c0.x; bfrag[nt][1] = c0.y; bfrag[nt][2] = c0.z; bfrag[nt][3] = c0.w;
            bfrag[nt][4] = c1.x; bfrag[nt][5] = c1.y; bfrag[nt][6] = c1.z; bfrag[nt][7] = c1.w;
        }
        #pragma unroll
        for (int mt = 0; mt < 4; ++mt)
            #pragma unroll
            for (int nt = 0; nt < 2; ++nt)
                acc[mt][nt] = __builtin_amdgcn_wmma_i32_16x16x64_iu8(
                     true, afrag[mt],  true, bfrag[nt],
                    acc[mt][nt],  false,  false);
    };

    const int KT = K / BK;
    stage(0, 0);
    int kt = 0;
    for (; kt < KT - 1; ++kt) {
        stage((kt + 1) * BK, (kt + 1) & 1);
        wait_async<4>();
        __syncthreads();
        compute(kt & 1);
        __syncthreads();
    }
    wait_async<0>();
    __syncthreads();
    compute(kt & 1);

    #pragma unroll
    for (int mt = 0; mt < 4; ++mt) {
        #pragma unroll
        for (int nt = 0; nt < 2; ++nt) {
            const int m0 = blockM + waveM + mt * 16 + half * 8;
            const int n  = blockN + waveN + nt * 16 + l16;
            const float bv = bias[n];
            #pragma unroll
            for (int r = 0; r < 8; ++r) {
                const int m = m0 + r;
                float v = (float)acc[mt][nt][r] * dqA[m] * dqB + bv;
                if (GELU) v = 0.5f * v * (1.0f + erff(v * 0.70710678118f));
                C[(size_t)m * N + n] = v;
            }
        }
    }
}

__global__ __launch_bounds__(256)
void ln_quant_kernel(const float* __restrict__ h, const float* __restrict__ gamma,
                     const float* __restrict__ beta, int8_t* __restrict__ hq,
                     float* __restrict__ dq) {
    __shared__ float red[256];
    __shared__ float red2[256];
    const int row = blockIdx.x;
    const int tid = threadIdx.x;
    const float* hr = h + (size_t)row * FDIM;

    float v[32];
    float sum = 0.0f, sq = 0.0f;
    #pragma unroll
    for (int j = 0; j < 32; ++j) {
        float t = hr[tid + j * 256];
        v[j] = t;
        sum += t;
        sq  += t * t;
    }
    red[tid] = sum; red2[tid] = sq;
    __syncthreads();
    for (int s2 = 128; s2 > 0; s2 >>= 1) {
        if (tid < s2) { red[tid] += red[tid + s2]; red2[tid] += red2[tid + s2]; }
        __syncthreads();
    }
    const float mean = red[0] * (1.0f / FDIM);
    const float var  = red2[0] * (1.0f / FDIM) - mean * mean;
    const float rstd = rsqrtf(var + EPSV);
    __syncthreads();

    float amax = 0.0f;
    #pragma unroll
    for (int j = 0; j < 32; ++j) {
        const int idx = tid + j * 256;
        float y = (v[j] - mean) * rstd * gamma[idx] + beta[idx];
        v[j] = y;
        amax = fmaxf(amax, fabsf(y));
    }
    red[tid] = amax;
    __syncthreads();
    for (int s2 = 128; s2 > 0; s2 >>= 1) {
        if (tid < s2) red[tid] = fmaxf(red[tid], red[tid + s2]);
        __syncthreads();
    }
    const float mx = fmaxf(red[0], EPSV);
    const float scale = 127.0f / mx;
    #pragma unroll
    for (int j = 0; j < 32; ++j) {
        float q = rintf(v[j] * scale);
        q = fminf(fmaxf(q, -128.0f), 127.0f);
        hq[(size_t)row * FDIM + tid + j * 256] = (int8_t)q;
    }
    if (tid == 0) dq[row] = mx / 127.0f;
}

extern "C" void kernel_launch(void* const* d_in, const int* in_sizes, int n_in,
                              void* d_out, int out_size, void* d_ws, size_t ws_size,
                              hipStream_t stream) {
    const float* x     = (const float*)d_in[0];
    const float* w1    = (const float*)d_in[1];
    const float* b1    = (const float*)d_in[2];
    const float* gamma = (const float*)d_in[3];
    const float* beta  = (const float*)d_in[4];
    const float* w2    = (const float*)d_in[5];
    const float* b2    = (const float*)d_in[6];
    float* out = (float*)d_out;

    char* ws = (char*)d_ws;
    size_t off = 0;
    int8_t* xq  = (int8_t*)(ws + off); off += (size_t)TOKENS * DDIM;
    int8_t* wq1 = (int8_t*)(ws + off); off += (size_t)WELEMS;
    int8_t* wq2 = (int8_t*)(ws + off); off += (size_t)WELEMS;
    int8_t* hq  = (int8_t*)(ws + off); off += (size_t)TOKENS * FDIM;
    off = (off + 255) & ~(size_t)255;
    float* h    = (float*)(ws + off);  off += (size_t)TOKENS * FDIM * 4;
    float* dqX  = (float*)(ws + off);  off += (size_t)TOKENS * 4;
    float* dqH  = (float*)(ws + off);  off += (size_t)TOKENS * 4;
    float* scal = (float*)(ws + off);  off += 256;

    zero_scalars_kernel<<<1, 32, 0, stream>>>(scal);
    abs_sum_kernel<<<2048, 256, 0, stream>>>(w1, (long long)WELEMS, scal + 0);
    abs_sum_kernel<<<2048, 256, 0, stream>>>(w2, (long long)WELEMS, scal + 1);
    make_scales_kernel<<<1, 32, 0, stream>>>(scal);
    quant_w_kernel<<<8192, 256, 0, stream>>>(w1, wq1, (long long)WELEMS, scal + 2);
    quant_w_kernel<<<8192, 256, 0, stream>>>(w2, wq2, (long long)WELEMS, scal + 3);

    quant_x_kernel<<<TOKENS, 256, 0, stream>>>(x, xq, dqX, DDIM);

    gemm_i8_kernel<true><<<dim3(FDIM / BN, TOKENS / BM), 256, 0, stream>>>(
        xq, wq1, dqX, scal, 2, b1, h, TOKENS, FDIM, DDIM);

    ln_quant_kernel<<<TOKENS, 256, 0, stream>>>(h, gamma, beta, hq, dqH);

    gemm_i8_kernel<false><<<dim3(DDIM / BN, TOKENS / BM), 256, 0, stream>>>(
        hq, wq2, dqH, scal, 3, b2, out, TOKENS, DDIM, FDIM);
}
